// QuantumImageDecoder_6957847020216
// MI455X (gfx1250) — matrix operand order repaired and checked by reading, not compiled, not run on hardware
//
#include <hip/hip_runtime.h>

// ---------------------------------------------------------------------------
// QuantumImageDecoder for MI455X (gfx1250, wave32, WMMA, TDM/async-to-LDS)
//   K0: zero fp32 h-partial buffer
//   K1: per-batch-row quantum gate butterflies entirely in LDS -> |psi|^2 (bf16)
//       (state row loaded via tensor_load_to_lds when TDM builtin available)
//   K2: GEMM1 split-K x8:  h_f32 += decoded @ W1^T   (global_atomic_add_f32)
//   K3: bias + relu + bf16 convert of h
//   K4: GEMM2: out = h @ W2^T + b2  (W2 streamed non-temporally, read once)
// ---------------------------------------------------------------------------

#define NQ      14
#define DIM     16384     // 2^14
#define BATCH   256
#define H1      512
#define NOUT    65536     // 1*256*256
#define KSPLIT1 8

typedef __attribute__((ext_vector_type(16))) __bf16       v16bf;
typedef __attribute__((ext_vector_type(8)))  float        v8f;
typedef __attribute__((ext_vector_type(4)))  float        v4f;
typedef __attribute__((ext_vector_type(4)))  unsigned int v4u;
typedef __attribute__((ext_vector_type(8)))  int          v8i;
typedef __attribute__((ext_vector_type(4)))  int          v4i;

union Frag32B { uint4 u[2]; v16bf v; };   // 32 bytes = 16 bf16 = one WMMA A/B fragment
union Pack8BF { __bf16 h[8]; uint4 u; };

#if __has_builtin(__builtin_amdgcn_tensor_load_to_lds) && __has_builtin(__builtin_amdgcn_s_wait_tensorcnt)
#define HAVE_TDM 1
#else
#define HAVE_TDM 0
#endif

#if __has_builtin(__builtin_amdgcn_global_load_async_to_lds_b128) && __has_builtin(__builtin_amdgcn_s_wait_asynccnt)
#define HAVE_ASYNC 1
#else
#define HAVE_ASYNC 0
#endif

// ---------------------------------------------------------------------------
// K0: zero the fp32 split-K accumulator (256*512 floats).
// ---------------------------------------------------------------------------
__global__ __launch_bounds__(256) void zero_kernel(float* __restrict__ p) {
  const int i = (blockIdx.x * 256 + threadIdx.x) * 4;
  *(v4f*)(p + i) = (v4f)0.0f;
}

// ---------------------------------------------------------------------------
// K1: quantum decode. grid = BATCH, block = 256, dyn LDS = 128 KB
// (re/im fp32 state; gfx1250 WGP has 320 KB LDS so 128 KB/WG is legal).
// ---------------------------------------------------------------------------
__global__ __launch_bounds__(256) void decode_kernel(const float* __restrict__ qs,
                                                     const float* __restrict__ params,
                                                     __bf16* __restrict__ decoded) {
  extern __shared__ float smem[];
  float* re = smem;
  float* im = smem + DIM;
  const int b   = blockIdx.x;
  const int tid = threadIdx.x;

#if HAVE_TDM
  // Tensor Data Mover: DMA one 64 KB state row (1 x 16384 f32 tile) into LDS.
  if (tid < 32) {  // wave 0 issues the descriptor (EXEC ignored by TDM)
    const unsigned long long ga = (unsigned long long)(const void*)(qs + (size_t)b * DIM);
    const unsigned int lds_off  = (unsigned int)(unsigned long long)re;
    v4u g0 = { 1u,                                   // count = 1 valid descriptor
               lds_off,                               // lds_addr
               (unsigned int)(ga & 0xffffffffull),    // global_addr[31:0]
               (unsigned int)((ga >> 32) & 0x01ffffffull) | 0x80000000u }; // [56:32] | type=2
    v8i g1 = { (int)0x00020000,    // data_size = 4B
               (int)0x40000000,    // tensor_dim0 = 16384 (low 16 in [31:16])
               (int)0x00010000,    // tensor_dim1 = 1
               (int)0x40000000,    // tile_dim0 = 16384
               (int)0x00000001,    // tile_dim1 = 1
               (int)16384,         // tensor_dim0_stride = 16384
               0, 0 };
    v4i g2 = { 0, 0, 0, 0 };
    v4i g3 = { 0, 0, 0, 0 };
#if defined(__clang_major__) && (__clang_major__ >= 23)
    v8i g4 = { 0, 0, 0, 0, 0, 0, 0, 0 };
    __builtin_amdgcn_tensor_load_to_lds(g0, g1, g2, g3, g4, 0);
#else
    __builtin_amdgcn_tensor_load_to_lds(g0, g1, g2, g3, 0);
#endif
    __builtin_amdgcn_s_wait_tensorcnt(0);
  }
  for (int i = tid; i < DIM; i += 256) im[i] = 0.0f;
  __syncthreads();
#else
  for (int i = tid; i < DIM; i += 256) {
    re[i] = qs[(size_t)b * DIM + i];
    im[i] = 0.0f;
  }
  __syncthreads();
#endif

  for (int q = 0; q < NQ; ++q) {
    // gate = Rx(hx) Ry(hy) Rz(hz), angles = -params * 0.5
    const float hx = -0.5f * params[q * 3 + 0];
    const float hy = -0.5f * params[q * 3 + 1];
    const float hz = -0.5f * params[q * 3 + 2];
    const float cx = cosf(hx), sx = sinf(hx);
    const float cy = cosf(hy), sy = sinf(hy);
    const float cz = cosf(hz), sz = sinf(hz);
    // g00 = (cx*cy - i sx*sy) * e^{-i hz};  g11 = conj(g00)
    // g10 = (cx*sy - i sx*cy) * e^{-i hz};  g01 = -conj(g10)
    const float g00r =  cx * cy * cz - sx * sy * sz;
    const float g00i = -cx * cy * sz - sx * sy * cz;
    const float g01r = -cx * sy * cz + sx * cy * sz;
    const float g01i = -cx * sy * sz - sx * cy * cz;
    const float g10r =  cx * sy * cz - sx * cy * sz;
    const float g10i = -cx * sy * sz - sx * cy * cz;
    const float g11r =  g00r;
    const float g11i = -g00i;

    const int sh    = 13 - q;      // right = 2^(13-q)
    const int right = 1 << sh;

    for (int t = tid; t < (DIM / 2); t += 256) {
      const int r  = t & (right - 1);
      const int l  = t >> sh;
      const int i0 = (l << (sh + 1)) | r;
      const int i1 = i0 + right;

      const float p0r = re[i0], p0i = im[i0];
      const float p1r = re[i1], p1i = im[i1];

      re[i0] = g00r * p0r - g00i * p0i + g01r * p1r - g01i * p1i;
      im[i0] = g00r * p0i + g00i * p0r + g01r * p1i + g01i * p1r;
      re[i1] = g10r * p0r - g10i * p0i + g11r * p1r - g11i * p1i;
      im[i1] = g10r * p0i + g10i * p0r + g11r * p1i + g11i * p1r;
    }
    __syncthreads();
  }

  for (int i = tid; i < DIM; i += 256) {
    const float p = re[i] * re[i] + im[i] * im[i];
    decoded[(size_t)b * DIM + i] = (__bf16)p;
  }
}

// ---------------------------------------------------------------------------
// Unified WMMA GEMM over a [kb, kb+KCH) K-range:
//   Out(256 x N)  (+)=  A(256 x KTOT, bf16) @ W(N x KTOT, f32)^T
// One workgroup covers the full M=256 and a 64-wide N strip.
// 8 waves, each owns a 64x32 tile (4x2 wmma accums of v8f).
// ATOMIC=true  -> split-K: global_atomic_add_f32 into fp32 Out
// ATOMIC=false -> bias (+relu) epilogue, direct store
// ---------------------------------------------------------------------------
template <int KTOT, int N, int KCH, bool ATOMIC, bool RELU, typename OutT>
__global__ __launch_bounds__(256) void gemm_wmma_kernel(const __bf16* __restrict__ A,
                                                        const float*  __restrict__ W,
                                                        const float*  __restrict__ bias,
                                                        OutT* __restrict__ Out) {
  __shared__ __attribute__((aligned(16))) __bf16 lds_a[256 * 32];  // [m][k] 16 KB
  __shared__ __attribute__((aligned(16))) __bf16 lds_b[64 * 32];   // [n][k]  4 KB

  const int tid   = threadIdx.x;
  const int n0    = blockIdx.x * 64;
  const int kb    = blockIdx.y * KCH;
  const int wave  = tid >> 5;
  const int lane  = tid & 31;
  const int wm    = wave >> 1;   // 0..3 -> 64-row M block
  const int wn    = wave & 1;    // 0..1 -> 32-col N block
  const int lhalf = lane >> 4;   // lane half per ISA fragment layout
  const int lidx  = lane & 15;

  v8f acc[4][2];
  for (int i = 0; i < 4; ++i)
    for (int j = 0; j < 2; ++j)
      for (int r = 0; r < 8; ++r) acc[i][j][r] = 0.0f;

  const int a_part = tid & 3;    // 16B chunk within a 64B A row
  const int b_n    = tid >> 2;   // 0..63
  const int b_part = tid & 3;    // 32B chunk within a 128B W row

  for (int k0 = kb; k0 < kb + KCH; k0 += 32) {
    // ---- stage A tile: 256 rows x 32 k of bf16 (pure copy -> async-to-LDS) ----
#if HAVE_ASYNC
    #pragma unroll
    for (int it = 0; it < 4; ++it) {
      const int row = (tid >> 2) + it * 64;
      __builtin_amdgcn_global_load_async_to_lds_b128(
          (__attribute__((address_space(1))) v4i*)(A + (size_t)row * KTOT + k0 + a_part * 8),
          (__attribute__((address_space(3))) v4i*)&lds_a[row * 32 + a_part * 8],
          0, 0);
    }
#else
    #pragma unroll
    for (int it = 0; it < 4; ++it) {
      const int row = (tid >> 2) + it * 64;
      const uint4 va = *(const uint4*)(A + (size_t)row * KTOT + k0 + a_part * 8);
      *(uint4*)&lds_a[row * 32 + a_part * 8] = va;
    }
#endif
    // ---- stage B tile: 64 n-rows x 32 k; stream W non-temporally, cvt to bf16 ----
    {
      const float* gw = W + (size_t)(n0 + b_n) * KTOT + k0 + b_part * 8;
      const v4f f0 = __builtin_nontemporal_load((const v4f*)(gw + 0));
      const v4f f1 = __builtin_nontemporal_load((const v4f*)(gw + 4));
      Pack8BF t;
      t.h[0] = (__bf16)f0.x; t.h[1] = (__bf16)f0.y;
      t.h[2] = (__bf16)f0.z; t.h[3] = (__bf16)f0.w;
      t.h[4] = (__bf16)f1.x; t.h[5] = (__bf16)f1.y;
      t.h[6] = (__bf16)f1.z; t.h[7] = (__bf16)f1.w;
      *(uint4*)&lds_b[b_n * 32 + b_part * 8] = t.u;
    }
    // prefetch next W strip into the cache hierarchy (global_prefetch_b8)
    if (k0 + 32 < kb + KCH)
      __builtin_prefetch((const void*)(W + (size_t)(n0 + b_n) * KTOT + k0 + 32 + b_part * 8), 0, 3);

#if HAVE_ASYNC
    __builtin_amdgcn_s_wait_asynccnt(0);
#endif
    __syncthreads();

    // ---- assemble fragments per ISA 7.12.2 layouts ----
    Frag32B fa[4];
    #pragma unroll
    for (int ms = 0; ms < 4; ++ms) {
      const int m = wm * 64 + ms * 16 + lidx;
      const __bf16* pa = &lds_a[m * 32 + 8 * lhalf];  // K chunks: 8*hi and 16+8*hi
      fa[ms].u[0] = *(const uint4*)(pa + 0);
      fa[ms].u[1] = *(const uint4*)(pa + 16);
    }
    Frag32B fb[2];
    #pragma unroll
    for (int ns = 0; ns < 2; ++ns) {
      const int n = wn * 32 + ns * 16 + lidx;
      const __bf16* pb = &lds_b[n * 32 + 8 * lhalf];  // K chunks: 8*hi and 16+8*hi
      fb[ns].u[0] = *(const uint4*)(pb + 0);
      fb[ns].u[1] = *(const uint4*)(pb + 16);
    }

    #pragma unroll
    for (int ms = 0; ms < 4; ++ms)
      #pragma unroll
      for (int ns = 0; ns < 2; ++ns)
        acc[ms][ns] = __builtin_amdgcn_wmma_f32_16x16x32_bf16(
            false, fa[ms].v, false, fb[ns].v, (short)0, acc[ms][ns], false, false);

    __syncthreads();
  }

  // ---- epilogue: scatter per C/D fragment layout ----
  (void)bias;
  #pragma unroll
  for (int ns = 0; ns < 2; ++ns) {
    const int n_g = n0 + wn * 32 + ns * 16 + lidx;
    if constexpr (ATOMIC) {
      #pragma unroll
      for (int ms = 0; ms < 4; ++ms)
        #pragma unroll
        for (int r = 0; r < 8; ++r) {
          const int m = wm * 64 + ms * 16 + r + 8 * lhalf;
          atomicAdd((float*)&Out[(size_t)m * N + n_g], acc[ms][ns][r]);
        }
    } else {
      const float bv = bias[n_g];
      #pragma unroll
      for (int ms = 0; ms < 4; ++ms) {
        #pragma unroll
        for (int r = 0; r < 8; ++r) {
          const int m = wm * 64 + ms * 16 + r + 8 * lhalf;
          float v = acc[ms][ns][r] + bv;
          if (RELU) v = fmaxf(v, 0.0f);
          Out[(size_t)m * N + n_g] = (OutT)v;
        }
      }
    }
  }
}

// ---------------------------------------------------------------------------
// K3: h_bf16 = bf16(relu(h_f32 + b1))  over 256 x 512
// ---------------------------------------------------------------------------
__global__ __launch_bounds__(256) void bias_relu_kernel(const float* __restrict__ hf,
                                                        const float* __restrict__ b1,
                                                        __bf16* __restrict__ h) {
  const int i = blockIdx.x * 256 + threadIdx.x;           // 0 .. 131071
  const float v = hf[i] + b1[i & (H1 - 1)];
  h[i] = (__bf16)fmaxf(v, 0.0f);
}

// ---------------------------------------------------------------------------
extern "C" void kernel_launch(void* const* d_in, const int* in_sizes, int n_in,
                              void* d_out, int out_size, void* d_ws, size_t ws_size,
                              hipStream_t stream) {
  const float* qs     = (const float*)d_in[0];  // (256, 16384)
  const float* params = (const float*)d_in[1];  // (1, 14, 3)
  const float* W1     = (const float*)d_in[2];  // (512, 16384)
  const float* b1     = (const float*)d_in[3];  // (512,)
  const float* W2     = (const float*)d_in[4];  // (65536, 512)
  const float* b2     = (const float*)d_in[5];  // (65536,)
  (void)in_sizes; (void)n_in; (void)out_size; (void)ws_size;

  __bf16* decoded = (__bf16*)d_ws;                                               // 8 MB
  float*  h_f32   = (float*)((char*)d_ws + (size_t)BATCH * DIM * 2);             // 512 KB
  __bf16* h_bf16  = (__bf16*)((char*)h_f32 + (size_t)BATCH * H1 * sizeof(float)); // 256 KB

  // K0: zero split-K accumulator (256*512 floats, 4/thread)
  zero_kernel<<<(BATCH * H1) / (256 * 4), 256, 0, stream>>>(h_f32);

  // K1: one WG per batch row, 128 KB dynamic LDS for the complex state.
  decode_kernel<<<BATCH, 256, 2 * DIM * sizeof(float), stream>>>(qs, params, decoded);

  // K2: split-K GEMM1 partials: h_f32 += decoded @ W1^T   (grid 8 x 8)
  gemm_wmma_kernel<DIM, H1, DIM / KSPLIT1, true, false, float>
      <<<dim3(H1 / 64, KSPLIT1), 256, 0, stream>>>(decoded, W1, b1, h_f32);

  // K3: bias + relu + convert to bf16
  bias_relu_kernel<<<(BATCH * H1) / 256, 256, 0, stream>>>(h_f32, b1, h_bf16);

  // K4: out = h @ W2^T + b2   (256 x 65536), W2 read exactly once
  gemm_wmma_kernel<H1, NOUT, H1, false, false, float>
      <<<dim3(NOUT / 64, 1), 256, 0, stream>>>(h_bf16, W2, b2, (float*)d_out);
}
